// BayesianLinear_83769042141907
// MI455X (gfx1250) — matrix operand order repaired and checked by reading, not compiled, not run on hardware
//
#include <hip/hip_runtime.h>

// ---------------------------------------------------------------------------
// BayesianLinear on gfx1250 (MI455X):
//   pass 1: W = w_mu + softplus(w_rho)*eps_w  -> bf16   (workspace)
//           Xb = bf16(x)                                (workspace)
//           b  = b_mu + softplus(b_rho)*eps_b -> f32    (workspace)
//   pass 2: Y = Xb @ Wb^T + b  via v_wmma_f32_16x16x32_bf16, f32 accum.
// bf16 X (32MB) + bf16 W (32MB) fit in the 192MB L2. GEMM uses 128x256
// block tiles, 8 waves, 64x64 per wave (4x4 fragments) so each 16-WMMA
// burst needs only 16 global_load_b128 (512B of operand fill per WMMA).
// ---------------------------------------------------------------------------

typedef __bf16 bf16;
typedef __attribute__((ext_vector_type(16))) __bf16 v16bf;
typedef __attribute__((ext_vector_type(8)))  __bf16 v8bf;
typedef __attribute__((ext_vector_type(8)))  float  v8f;

static __device__ __forceinline__ unsigned short f32_to_bf16_rne(float f) {
  unsigned int u = __float_as_uint(f);
  u += 0x7FFFu + ((u >> 16) & 1u);   // round-to-nearest-even
  return (unsigned short)(u >> 16);
}

static __device__ __forceinline__ float softplus_f(float x) {
  // numerically stable log1p(exp(x))
  return fmaxf(x, 0.0f) + log1pf(expf(-fabsf(x)));
}

// ---- prep: W = mu + softplus(rho)*eps, packed to bf16 (4 elems/thread) ----
__global__ void bayes_prep_w(const float4* __restrict__ mu,
                             const float4* __restrict__ rho,
                             const float4* __restrict__ eps,
                             ushort4* __restrict__ wq, int n4) {
  int i = blockIdx.x * blockDim.x + threadIdx.x;
  if (i >= n4) return;
  float4 m = mu[i], r = rho[i], e = eps[i];
  ushort4 o;
  o.x = f32_to_bf16_rne(fmaf(softplus_f(r.x), e.x, m.x));
  o.y = f32_to_bf16_rne(fmaf(softplus_f(r.y), e.y, m.y));
  o.z = f32_to_bf16_rne(fmaf(softplus_f(r.z), e.z, m.z));
  o.w = f32_to_bf16_rne(fmaf(softplus_f(r.w), e.w, m.w));
  wq[i] = o;
}

// ---- prep: X -> bf16 ----
__global__ void bayes_prep_x(const float4* __restrict__ x,
                             ushort4* __restrict__ xq, int n4) {
  int i = blockIdx.x * blockDim.x + threadIdx.x;
  if (i >= n4) return;
  float4 v = x[i];
  ushort4 o;
  o.x = f32_to_bf16_rne(v.x);
  o.y = f32_to_bf16_rne(v.y);
  o.z = f32_to_bf16_rne(v.z);
  o.w = f32_to_bf16_rne(v.w);
  xq[i] = o;
}

// ---- prep: b = b_mu + softplus(b_rho)*eps_b (f32) ----
__global__ void bayes_prep_b(const float* __restrict__ bmu,
                             const float* __restrict__ brho,
                             const float* __restrict__ beps,
                             float* __restrict__ b, int n) {
  int i = blockIdx.x * blockDim.x + threadIdx.x;
  if (i < n) b[i] = fmaf(softplus_f(brho[i]), beps[i], bmu[i]);
}

// A fragment loader: two 16B chunks (K..K+7 halves in regs 0..3,
// K+8..K+15 halves in regs 4..7), contiguous K to match B.
static __device__ __forceinline__ v16bf load_afrag(const char* base, unsigned off) {
  v8bf lo = *(const v8bf*)(base + off);
  v8bf hi = *(const v8bf*)(base + off + 16);
  return __builtin_shufflevector(lo, hi,
      0, 1, 2, 3, 4, 5, 6, 7, 8, 9, 10, 11, 12, 13, 14, 15);
}

// B fragment loader: one contiguous 32B chunk (16 consecutive K halves).
static __device__ __forceinline__ v16bf load_bfrag(const char* base, unsigned off) {
  return *(const v16bf*)(base + off);
}

// ---------------------------------------------------------------------------
// WMMA GEMM: Y[m,n] = sum_k Xb[m,k]*Wb[n,k] + bias[n]
// Block: 128(M) x 256(N), 256 threads = 8 waves arranged 2(M) x 4(N).
// Each wave: 64x64 output = 4(M) x 4(N) fragments of 16x16, K step 32.
// Loads are issued at the top of each iteration; the compiler's scheduler
// hoists the next iteration's loads across the 16-WMMA burst (verified in
// the round-2 asm) so loads stay in flight during compute.
//
// wave32 operand layouts (CDNA5 ISA 7.12.2):
//   A 16x32 bf16: lane<16 -> M=lane, halves 0..7 = K(kk + 0..7),
//                 halves 8..15 = K(kk + 16..23); lane>=16 -> K ranges +8.
//   B 32x16 bf16 (code's order, A same, not ISA's): lane<16 -> N=lane, halves 0..15 = K(kk + 0..15);
//                 lane>=16 -> K(kk + 16..31). Row-major W == this layout.
//   C/D f32: VGPR v -> row v + 8*(lane>=16), col lane%16.
// ---------------------------------------------------------------------------
__launch_bounds__(256)
__global__ void bayes_wmma_gemm(const bf16* __restrict__ X,
                                const bf16* __restrict__ W,
                                const float* __restrict__ bias,
                                float* __restrict__ Y,
                                int K, int Nout) {
  const int lane  = threadIdx.x & 31;
  const int wave  = threadIdx.x >> 5;
  const int wm    = wave >> 2;      // 0..1 (M direction)
  const int wn    = wave & 3;      // 0..3 (N direction)
  const int lhalf = lane >> 4;      // 0/1
  const int l16   = lane & 15;

  const int bm = blockIdx.y * 128;
  const int bn = blockIdx.x * 256;

  const int arow0 = bm + wm * 64 + l16;   // A fragment base row (per lane)
  const int bcol0 = bn + wn * 64 + l16;   // B fragment base col (per lane)

  // Loop-invariant per-lane byte offsets; uniform bases advance 64B / 32-K.
  unsigned aoff[4], boff[4];
#pragma unroll
  for (int mf = 0; mf < 4; ++mf)
    aoff[mf] = ((unsigned)(arow0 + mf * 16) * (unsigned)K + (unsigned)(lhalf * 16)) * 2u;
#pragma unroll
  for (int nf = 0; nf < 4; ++nf)
    boff[nf] = ((unsigned)(bcol0 + nf * 16) * (unsigned)K + (unsigned)(lhalf * 16)) * 2u;

  const char* xk = (const char*)X;
  const char* wk = (const char*)W;

  v8f acc[4][4];
#pragma unroll
  for (int mf = 0; mf < 4; ++mf) {
#pragma unroll
    for (int nf = 0; nf < 4; ++nf) acc[mf][nf] = 0.0f;
  }

  for (int kk = 0; kk < K; kk += 32) {
    v16bf afrag[4], bfrag[4];
#pragma unroll
    for (int mf = 0; mf < 4; ++mf) afrag[mf] = load_afrag(xk, aoff[mf]);
#pragma unroll
    for (int nf = 0; nf < 4; ++nf) bfrag[nf] = load_bfrag(wk, boff[nf]);

#pragma unroll
    for (int mf = 0; mf < 4; ++mf) {
#pragma unroll
      for (int nf = 0; nf < 4; ++nf) {
        acc[mf][nf] = __builtin_amdgcn_wmma_f32_16x16x32_bf16(
            false, afrag[mf], false, bfrag[nf], (short)0, acc[mf][nf], false, false);
      }
    }

    xk += 64;
    wk += 64;
  }

  // --- epilogue: add bias, store f32 ---
#pragma unroll
  for (int nf = 0; nf < 4; ++nf) {
    const int col = bcol0 + nf * 16;     // per-lane output column
    const float bv = bias[col];
#pragma unroll
    for (int mf = 0; mf < 4; ++mf) {
      const int row0 = bm + wm * 64 + mf * 16 + lhalf * 8;
      float* yp = Y + (size_t)row0 * Nout + col;
#pragma unroll
      for (int v = 0; v < 8; ++v) {
        yp[(size_t)v * Nout] = acc[mf][nf][v] + bv;
      }
    }
  }
}

extern "C" void kernel_launch(void* const* d_in, const int* in_sizes, int n_in,
                              void* d_out, int out_size, void* d_ws, size_t ws_size,
                              hipStream_t stream) {
  const float* x     = (const float*)d_in[0];
  const float* w_mu  = (const float*)d_in[1];
  const float* w_rho = (const float*)d_in[2];
  const float* b_mu  = (const float*)d_in[3];
  const float* b_rho = (const float*)d_in[4];
  const float* eps_w = (const float*)d_in[5];
  const float* eps_b = (const float*)d_in[6];

  const int   OUT_F = in_sizes[3];                   // b_mu length
  const long long WN = (long long)in_sizes[1];       // OUT_F * IN_F
  const int   IN_F  = (int)(WN / OUT_F);
  const long long XN = (long long)in_sizes[0];       // N_TOK * IN_F
  const int   N_TOK = (int)(XN / IN_F);

  // workspace: [ Xb bf16 | Wb bf16 | bias f32 ]
  char* ws = (char*)d_ws;
  bf16*  Xb = (bf16*)ws;
  bf16*  Wb = (bf16*)(ws + XN * sizeof(bf16));
  float* bb = (float*)(ws + XN * sizeof(bf16) + WN * sizeof(bf16));

  {
    int n4 = (int)(WN / 4);
    bayes_prep_w<<<(n4 + 255) / 256, 256, 0, stream>>>(
        (const float4*)w_mu, (const float4*)w_rho, (const float4*)eps_w,
        (ushort4*)Wb, n4);
  }
  {
    int n4 = (int)(XN / 4);
    bayes_prep_x<<<(n4 + 255) / 256, 256, 0, stream>>>(
        (const float4*)x, (ushort4*)Xb, n4);
  }
  {
    bayes_prep_b<<<(OUT_F + 255) / 256, 256, 0, stream>>>(
        b_mu, b_rho, eps_b, bb, OUT_F);
  }

  dim3 grid(OUT_F / 256, N_TOK / 128);
  bayes_wmma_gemm<<<grid, 256, 0, stream>>>(Xb, Wb, bb, (float*)d_out,
                                            IN_F, OUT_F);
}
